// ST_LLM_topk_memory_nog2_78202764525975
// MI455X (gfx1250) — compile-verified, matrix operand order checked by reading, not run on hardware
//
#include <hip/hip_runtime.h>
#include <math.h>

#define USE_ASYNC_LDS 1

// ---------------------------------------------------------------------------
// Types for CDNA5 (gfx1250) WMMA: wave32, v_wmma_f32_16x16x32_bf16
// ---------------------------------------------------------------------------
typedef __attribute__((ext_vector_type(16))) __bf16        v16bf;
typedef __attribute__((ext_vector_type(8)))  float         v8f;
typedef __attribute__((ext_vector_type(4)))  unsigned int  u32x4;

union FragAB {
    v16bf v;
    u32x4 u[2];
};

__device__ __forceinline__ unsigned short f2bf(float f) {
    unsigned u = __float_as_uint(f);
    unsigned r = u + 0x7FFFu + ((u >> 16) & 1u);   // round-to-nearest-even
    return (unsigned short)(r >> 16);
}

__device__ __forceinline__ float wredsum(float v) {
#pragma unroll
    for (int o = 16; o > 0; o >>= 1) v += __shfl_xor(v, o, 32);
    return v;
}

__device__ __forceinline__ float gelu_exact(float v) {
    return 0.5f * v * (1.0f + erff(v * 0.7071067811865476f));
}

#if USE_ASYNC_LDS
// gfx1250 async copy global -> LDS (16B per lane), tracked with ASYNCcnt.
// %0 = LDS byte address (VGPR), %1 = 64-bit global address (VGPR pair).
__device__ __forceinline__ void async_ld_b128(unsigned lds_off, const void* gaddr) {
    asm volatile("global_load_async_to_lds_b128 %0, %1, off"
                 :: "v"(lds_off), "v"(gaddr) : "memory");
}
__device__ __forceinline__ void wait_async0() {
    asm volatile("s_wait_asynccnt 0" ::: "memory");
}
// Flat shared address truncates to LDS byte offset.
__device__ __forceinline__ unsigned lds_addr(const void* p) {
    return (unsigned)(size_t)p;
}
#endif

// ---------------------------------------------------------------------------
// adj normalization: dinv = (rowsum(adj)+1)^-0.5 ; adjn = dinv_i*(a+I)*dinv_j
// ---------------------------------------------------------------------------
__global__ void k_dinv(const float* __restrict__ adj, float* __restrict__ dinv, int N) {
    int wid = threadIdx.x >> 5, lane = threadIdx.x & 31;
    int row = blockIdx.x * 8 + wid;
    if (row >= N) return;
    float s = 0.f;
    for (int j = lane; j < N; j += 32) s += adj[(size_t)row * N + j];
    s = wredsum(s) + 1.0f;                         // + identity diagonal
    if (lane == 0) dinv[row] = (s > 0.f) ? rsqrtf(s) : 0.f;
}

__global__ void k_adjnorm(const float* __restrict__ adj, const float* __restrict__ dinv,
                          unsigned short* __restrict__ out, int N) {
    int idx = blockIdx.x * blockDim.x + threadIdx.x;
    if (idx >= N * N) return;
    int i = idx / N, j = idx - i * N;
    float v = adj[idx] + ((i == j) ? 1.f : 0.f);
    out[idx] = f2bf(v * dinv[i] * dinv[j]);
}

// ---------------------------------------------------------------------------
// fp32 -> bf16 transpose (builds Bt = [Ncols][K] operands)
// ---------------------------------------------------------------------------
__global__ void k_f2bf_T(const float* __restrict__ src, unsigned short* __restrict__ dst,
                         int R, int C) {
    size_t idx = (size_t)blockIdx.x * blockDim.x + threadIdx.x;
    if (idx >= (size_t)R * C) return;
    size_t r = idx / C, c = idx - r * C;
    dst[c * (size_t)R + r] = f2bf(src[idx]);
}

// fp32 -> bf16 with row stride (x into cat[:, colOff:colOff+C])
__global__ void k_f2bf_stride(const float* __restrict__ src, unsigned short* __restrict__ dst,
                              int C, int dstStride, int colOff, long long total) {
    size_t idx = (size_t)blockIdx.x * blockDim.x + threadIdx.x;
    if (idx >= (size_t)total) return;
    size_t r = idx / C;
    int c = (int)(idx - r * C);
    dst[r * dstStride + colOff + c] = f2bf(src[idx]);
}

// ---------------------------------------------------------------------------
// Tiled WMMA GEMM:  C[M,N] (fp32) = A_bf16[M,K] * Bt_bf16[N,K]^T  (+bias, act)
// 128x128 workgroup tile, 256 threads = 8 wave32 waves (4 in M x 2 in N),
// each wave 32x64 via 2x4 grid of v_wmma_f32_16x16x32_bf16.
// Global->LDS staging uses gfx1250 async copy (ASYNCcnt), double-buffered.
// LDS rows padded to 56 shorts (112B): 16B-aligned b128 access, conflict-free.
// ---------------------------------------------------------------------------
#define LDSP 56

__global__ __launch_bounds__(256) void k_gemm(
    const unsigned short* __restrict__ A, int lda,
    const unsigned short* __restrict__ Bt, int ldb,
    float* __restrict__ C, int ldc, int K,
    const float* __restrict__ bias, int act)
{
    __shared__ unsigned short As[2][128 * LDSP];
    __shared__ unsigned short Bs[2][128 * LDSP];

    const int t    = threadIdx.x;
    const int m0   = blockIdx.y * 128;
    const int n0   = blockIdx.x * 128;
    const int lr   = t >> 2;            // 0..63 (stages rows lr and lr+64)
    const int lc   = (t & 3) * 8;       // col in shorts: 0,8,16,24
    const int wid  = t >> 5;
    const int lane = t & 31;
    const int wm   = (wid & 3) * 32;    // wave M offset in tile
    const int wn   = (wid >> 2) * 64;   // wave N offset in tile
    const int frow = lane & 15;         // fragment row (A-layout)
    const int kb   = (lane >> 4) * 8;   // fragment K base (shorts)

    const unsigned short* agp0 = A  + (size_t)(m0 + lr)      * lda + lc;
    const unsigned short* agp1 = A  + (size_t)(m0 + lr + 64) * lda + lc;
    const unsigned short* bgp0 = Bt + (size_t)(n0 + lr)      * ldb + lc;
    const unsigned short* bgp1 = Bt + (size_t)(n0 + lr + 64) * ldb + lc;

    v8f acc[2][4];
#pragma unroll
    for (int i = 0; i < 2; ++i)
#pragma unroll
        for (int j = 0; j < 4; ++j)
#pragma unroll
            for (int r = 0; r < 8; ++r) acc[i][j][r] = 0.f;

    const int nk = K >> 5;

#if USE_ASYNC_LDS
    const unsigned la0 = lds_addr(&As[0][lr * LDSP + lc]);
    const unsigned la1 = lds_addr(&As[0][(lr + 64) * LDSP + lc]);
    const unsigned lb0 = lds_addr(&Bs[0][lr * LDSP + lc]);
    const unsigned lb1 = lds_addr(&Bs[0][(lr + 64) * LDSP + lc]);
    const unsigned bufB = (unsigned)(128 * LDSP * sizeof(unsigned short));

    // prologue: stage K-slice 0 into buffer 0 via async DMA
    async_ld_b128(la0, agp0);
    async_ld_b128(la1, agp1);
    async_ld_b128(lb0, bgp0);
    async_ld_b128(lb1, bgp1);
    wait_async0();
    __syncthreads();

    for (int kk = 0; kk < nk; ++kk) {
        const int cur = kk & 1;
        if (kk + 1 < nk) {              // async-stage next K-slice into nxt buffer
            const int k0 = (kk + 1) << 5;
            const unsigned nb = (unsigned)(cur ^ 1) * bufB;
            async_ld_b128(la0 + nb, agp0 + k0);
            async_ld_b128(la1 + nb, agp1 + k0);
            async_ld_b128(lb0 + nb, bgp0 + k0);
            async_ld_b128(lb1 + nb, bgp1 + k0);
        }
        if (kk + 2 < nk) {              // gfx1250 global_prefetch_b8 of slice k+2
            const int k2 = (kk + 2) << 5;
            __builtin_prefetch(agp0 + k2, 0, 1);
            __builtin_prefetch(bgp0 + k2, 0, 1);
        }

        FragAB a[2], b[4];
#pragma unroll
        for (int i = 0; i < 2; ++i) {
            const unsigned short* p = &As[cur][(wm + i * 16 + frow) * LDSP + kb];
            a[i].u[0] = *(const u32x4*)p;
            a[i].u[1] = *(const u32x4*)(p + 16);
        }
#pragma unroll
        for (int j = 0; j < 4; ++j) {
            const unsigned short* p = &Bs[cur][(wn + j * 16 + frow) * LDSP + kb];
            b[j].u[0] = *(const u32x4*)p;
            b[j].u[1] = *(const u32x4*)(p + 16);
        }
#pragma unroll
        for (int i = 0; i < 2; ++i)
#pragma unroll
            for (int j = 0; j < 4; ++j)
                acc[i][j] = __builtin_amdgcn_wmma_f32_16x16x32_bf16(
                    false, a[i].v, false, b[j].v, (short)0, acc[i][j], false, false);

        if (kk + 1 < nk) {
            wait_async0();              // this wave's DMA into nxt is done
            __syncthreads();            // everyone's DMA visible; cur free for reuse
        }
    }
#else
    u32x4 ra0, ra1, rb0, rb1;
    ra0 = *(const u32x4*)agp0; ra1 = *(const u32x4*)agp1;
    rb0 = *(const u32x4*)bgp0; rb1 = *(const u32x4*)bgp1;
    *(u32x4*)&As[0][lr * LDSP + lc]        = ra0;
    *(u32x4*)&As[0][(lr + 64) * LDSP + lc] = ra1;
    *(u32x4*)&Bs[0][lr * LDSP + lc]        = rb0;
    *(u32x4*)&Bs[0][(lr + 64) * LDSP + lc] = rb1;
    __syncthreads();
    for (int kk = 0; kk < nk; ++kk) {
        const int cur = kk & 1;
        if (kk + 1 < nk) {
            const int k0 = (kk + 1) << 5;
            ra0 = *(const u32x4*)(agp0 + k0); ra1 = *(const u32x4*)(agp1 + k0);
            rb0 = *(const u32x4*)(bgp0 + k0); rb1 = *(const u32x4*)(bgp1 + k0);
        }
        FragAB a[2], b[4];
#pragma unroll
        for (int i = 0; i < 2; ++i) {
            const unsigned short* p = &As[cur][(wm + i * 16 + frow) * LDSP + kb];
            a[i].u[0] = *(const u32x4*)p;
            a[i].u[1] = *(const u32x4*)(p + 16);
        }
#pragma unroll
        for (int j = 0; j < 4; ++j) {
            const unsigned short* p = &Bs[cur][(wn + j * 16 + frow) * LDSP + kb];
            b[j].u[0] = *(const u32x4*)p;
            b[j].u[1] = *(const u32x4*)(p + 16);
        }
#pragma unroll
        for (int i = 0; i < 2; ++i)
#pragma unroll
            for (int j = 0; j < 4; ++j)
                acc[i][j] = __builtin_amdgcn_wmma_f32_16x16x32_bf16(
                    false, a[i].v, false, b[j].v, (short)0, acc[i][j], false, false);
        if (kk + 1 < nk) {
            const int nxt = cur ^ 1;
            *(u32x4*)&As[nxt][lr * LDSP + lc]        = ra0;
            *(u32x4*)&As[nxt][(lr + 64) * LDSP + lc] = ra1;
            *(u32x4*)&Bs[nxt][lr * LDSP + lc]        = rb0;
            *(u32x4*)&Bs[nxt][(lr + 64) * LDSP + lc] = rb1;
            __syncthreads();
        }
    }
#endif

    // epilogue: C/D layout — VGPR r: lanes0-15 M=r, lanes16-31 M=8+r; N=lane&15
    const int mloc = 8 * (lane >> 4);
    const int ncol = lane & 15;
#pragma unroll
    for (int i = 0; i < 2; ++i)
#pragma unroll
        for (int j = 0; j < 4; ++j) {
            const int n = n0 + wn + j * 16 + ncol;
            const float bv = bias ? bias[n] : 0.f;
#pragma unroll
            for (int r = 0; r < 8; ++r) {
                const int m = m0 + wm + i * 16 + mloc + r;
                float v = acc[i][j][r] + bv;
                if (act) v = gelu_exact(v);
                C[(size_t)m * ldc + n] = v;
            }
        }
}

// ---------------------------------------------------------------------------
// Row L2-normalize (D=256) fp32 -> bf16 ; one wave per row, 8 rows per block
// ---------------------------------------------------------------------------
__global__ void k_rownorm(const float* __restrict__ src, unsigned short* __restrict__ dst, int R) {
    int wid = threadIdx.x >> 5, lane = threadIdx.x & 31;
    int row = blockIdx.x * 8 + wid;
    if (row >= R) return;
    const float4* p = (const float4*)(src + (size_t)row * 256 + lane * 8);
    float4 v0 = p[0], v1 = p[1];
    float ss = v0.x * v0.x + v0.y * v0.y + v0.z * v0.z + v0.w * v0.w +
               v1.x * v1.x + v1.y * v1.y + v1.z * v1.z + v1.w * v1.w;
    ss = wredsum(ss);
    float s = 1.0f / fmaxf(sqrtf(ss), 1e-12f);
    unsigned short* d = dst + (size_t)row * 256 + lane * 8;
    d[0] = f2bf(v0.x * s); d[1] = f2bf(v0.y * s); d[2] = f2bf(v0.z * s); d[3] = f2bf(v0.w * s);
    d[4] = f2bf(v1.x * s); d[5] = f2bf(v1.y * s); d[6] = f2bf(v1.z * s); d[7] = f2bf(v1.w * s);
}

// ---------------------------------------------------------------------------
// Per-node similarity: sim[b,n,m] = qn[b,n,:]·kn[n,m,:]  (32x32x256 WMMA)
// One wave32 per node n; EXEC all ones (WMMA requirement).
// ---------------------------------------------------------------------------
#define SIMP 280   // padded LDS row stride (shorts), 560B = 35*16

__global__ void k_sim(const unsigned short* __restrict__ qn,   // [B*N, 256]
                      const unsigned short* __restrict__ kn,   // [N*M, 256]
                      float* __restrict__ sim,                 // [B, N, M]
                      int N) {
    __shared__ unsigned short Q [32 * SIMP];
    __shared__ unsigned short Kt[32 * SIMP];
    const int n = blockIdx.x;
    const int lane = threadIdx.x;           // blockDim = 32
    for (int b = 0; b < 32; ++b) {
        *(u32x4*)&Q [b * SIMP + lane * 8] = *(const u32x4*)(qn + ((size_t)b * N + n)  * 256 + lane * 8);
        *(u32x4*)&Kt[b * SIMP + lane * 8] = *(const u32x4*)(kn + ((size_t)n * 32 + b) * 256 + lane * 8);
    }
    __syncthreads();

    const int frow = lane & 15, kb = (lane >> 4) * 8;
    v8f acc[2][2];
#pragma unroll
    for (int i = 0; i < 2; ++i)
#pragma unroll
        for (int j = 0; j < 2; ++j)
#pragma unroll
            for (int r = 0; r < 8; ++r) acc[i][j][r] = 0.f;

    for (int ks = 0; ks < 8; ++ks) {
        FragAB a[2], b[2];
#pragma unroll
        for (int i = 0; i < 2; ++i) {
            const unsigned short* p = &Q[(i * 16 + frow) * SIMP + ks * 32 + kb];
            a[i].u[0] = *(const u32x4*)p;
            a[i].u[1] = *(const u32x4*)(p + 16);
        }
#pragma unroll
        for (int j = 0; j < 2; ++j) {
            const unsigned short* p = &Kt[(j * 16 + frow) * SIMP + ks * 32 + kb];
            b[j].u[0] = *(const u32x4*)p;
            b[j].u[1] = *(const u32x4*)(p + 16);
        }
#pragma unroll
        for (int i = 0; i < 2; ++i)
#pragma unroll
            for (int j = 0; j < 2; ++j)
                acc[i][j] = __builtin_amdgcn_wmma_f32_16x16x32_bf16(
                    false, a[i].v, false, b[j].v, (short)0, acc[i][j], false, false);
    }
    const int mloc = 8 * (lane >> 4), nc = lane & 15;
#pragma unroll
    for (int i = 0; i < 2; ++i)
#pragma unroll
        for (int j = 0; j < 2; ++j)
#pragma unroll
            for (int r = 0; r < 8; ++r) {
                const int b = i * 16 + mloc + r;
                const int m = j * 16 + nc;
                sim[((size_t)b * N + n) * 32 + m] = acc[i][j][r];
            }
}

// ---------------------------------------------------------------------------
// Top-4 (over M=32) + softmax(T=0.7) + weighted gather of value rows.
// One wave per (b,n); writes fp32 mem_out and bf16 into cat at colOff.
// ---------------------------------------------------------------------------
__global__ void k_topk(const float* __restrict__ sim,    // [BN, 32]
                       const float* __restrict__ vals,   // [N*M, 256]
                       float* __restrict__ memout,       // [BN, 256]
                       unsigned short* __restrict__ cat, int catOff,
                       int N) {
    const int wid = threadIdx.x >> 5, lane = threadIdx.x & 31;
    const int rid = blockIdx.x * 8 + wid;       // rid = b*N + n
    const int n = rid & (N - 1);                // N = 1024 (pow2)

    float s = sim[(size_t)rid * 32 + lane];
    float tv[4]; int ti[4];
    float cv = s;
#pragma unroll
    for (int j = 0; j < 4; ++j) {
        float v = cv; int idx = lane;
#pragma unroll
        for (int o = 16; o > 0; o >>= 1) {
            float ov = __shfl_xor(v, o, 32);
            int   oi = __shfl_xor(idx, o, 32);
            if (ov > v || (ov == v && oi < idx)) { v = ov; idx = oi; }
        }
        tv[j] = v; ti[j] = idx;
        if (lane == idx) cv = -3.0e38f;
    }
    const float Tinv = 1.0f / 0.7f;
    float mx = tv[0];
    float e[4], se = 0.f;
#pragma unroll
    for (int j = 0; j < 4; ++j) { e[j] = expf((tv[j] - mx) * Tinv); se += e[j]; }
    const float inv = 1.0f / se;

    const int d = lane * 8;
    float o[8];
#pragma unroll
    for (int k = 0; k < 8; ++k) o[k] = 0.f;
#pragma unroll
    for (int j = 0; j < 4; ++j) {
        const float aj = e[j] * inv;
        const float4* vp = (const float4*)(vals + ((size_t)n * 32 + ti[j]) * 256 + d);
        float4 x0 = vp[0], x1 = vp[1];
        o[0] += aj * x0.x; o[1] += aj * x0.y; o[2] += aj * x0.z; o[3] += aj * x0.w;
        o[4] += aj * x1.x; o[5] += aj * x1.y; o[6] += aj * x1.z; o[7] += aj * x1.w;
    }
    float* mo = memout + (size_t)rid * 256 + d;
    unsigned short* cp = cat + (size_t)rid * 768 + catOff + d;
#pragma unroll
    for (int k = 0; k < 8; ++k) { mo[k] = o[k]; cp[k] = f2bf(o[k]); }
}

// ---------------------------------------------------------------------------
// Fusion: w = softmax(h @ f2W + f2b); fused = w0*x + w1*selfm + w2*nbrm (bf16)
// ---------------------------------------------------------------------------
__global__ void k_fusion(const float* __restrict__ h,
                         const float* __restrict__ f2W, const float* __restrict__ f2b,
                         const float* __restrict__ x,  const float* __restrict__ sm,
                         const float* __restrict__ nm,
                         unsigned short* __restrict__ fusedbf) {
    const int wid = threadIdx.x >> 5, lane = threadIdx.x & 31;
    const size_t rid = (size_t)blockIdx.x * 8 + wid;
    float l0 = 0.f, l1 = 0.f, l2 = 0.f;
#pragma unroll
    for (int i = 0; i < 8; ++i) {
        const int c = lane + 32 * i;
        const float hv = h[rid * 256 + c];
        l0 += hv * f2W[c * 3 + 0];
        l1 += hv * f2W[c * 3 + 1];
        l2 += hv * f2W[c * 3 + 2];
    }
    l0 = wredsum(l0) + f2b[0];
    l1 = wredsum(l1) + f2b[1];
    l2 = wredsum(l2) + f2b[2];
    const float mx = fmaxf(l0, fmaxf(l1, l2));
    const float e0 = expf(l0 - mx), e1 = expf(l1 - mx), e2 = expf(l2 - mx);
    const float inv = 1.0f / (e0 + e1 + e2);
    const float w0 = e0 * inv, w1 = e1 * inv, w2 = e2 * inv;

    const int d = lane * 8;
    const float4* xp = (const float4*)(x  + rid * 256 + d);
    const float4* sp = (const float4*)(sm + rid * 256 + d);
    const float4* np = (const float4*)(nm + rid * 256 + d);
    float4 xa = xp[0], xb = xp[1], sa = sp[0], sb = sp[1], na = np[0], nb = np[1];
    unsigned short* fp = fusedbf + rid * 256 + d;
    fp[0] = f2bf(w0 * xa.x + w1 * sa.x + w2 * na.x);
    fp[1] = f2bf(w0 * xa.y + w1 * sa.y + w2 * na.y);
    fp[2] = f2bf(w0 * xa.z + w1 * sa.z + w2 * na.z);
    fp[3] = f2bf(w0 * xa.w + w1 * sa.w + w2 * na.w);
    fp[4] = f2bf(w0 * xb.x + w1 * sb.x + w2 * nb.x);
    fp[5] = f2bf(w0 * xb.y + w1 * sb.y + w2 * nb.y);
    fp[6] = f2bf(w0 * xb.z + w1 * sb.z + w2 * nb.z);
    fp[7] = f2bf(w0 * xb.w + w1 * sb.w + w2 * nb.w);
}

// ---------------------------------------------------------------------------
// Residual + LayerNorm (D=256), one wave per row
// ---------------------------------------------------------------------------
__global__ void k_lnres(const float* __restrict__ x, const float* __restrict__ ob,
                        const float* __restrict__ g, const float* __restrict__ be,
                        float* __restrict__ out) {
    const int wid = threadIdx.x >> 5, lane = threadIdx.x & 31;
    const size_t rid = (size_t)blockIdx.x * 8 + wid;
    const int d = lane * 8;
    const float4* xp = (const float4*)(x  + rid * 256 + d);
    const float4* op = (const float4*)(ob + rid * 256 + d);
    float4 xa = xp[0], xb = xp[1], oa = op[0], obv = op[1];
    float y[8] = { xa.x + oa.x,  xa.y + oa.y,  xa.z + oa.z,  xa.w + oa.w,
                   xb.x + obv.x, xb.y + obv.y, xb.z + obv.z, xb.w + obv.w };
    float s = 0.f;
#pragma unroll
    for (int k = 0; k < 8; ++k) s += y[k];
    const float mu = wredsum(s) * (1.0f / 256.0f);
    float vs = 0.f;
#pragma unroll
    for (int k = 0; k < 8; ++k) { const float t = y[k] - mu; vs += t * t; }
    const float var = wredsum(vs) * (1.0f / 256.0f);
    const float rstd = rsqrtf(var + 1e-5f);
    float* po = out + rid * 256 + d;
#pragma unroll
    for (int k = 0; k < 8; ++k)
        po[k] = (y[k] - mu) * rstd * g[d + k] + be[d + k];
}

// ---------------------------------------------------------------------------
// Host orchestration
// ---------------------------------------------------------------------------
extern "C" void kernel_launch(void* const* d_in, const int* in_sizes, int n_in,
                              void* d_out, int out_size, void* d_ws, size_t ws_size,
                              hipStream_t stream) {
    const float* x     = (const float*)d_in[0];
    const float* adj   = (const float*)d_in[1];
    const float* mkeys = (const float*)d_in[2];
    const float* mvals = (const float*)d_in[3];
    const float* qW    = (const float*)d_in[4];
    const float* qb    = (const float*)d_in[5];
    const float* f1W   = (const float*)d_in[6];
    const float* f1b   = (const float*)d_in[7];
    const float* f2W   = (const float*)d_in[8];
    const float* f2b   = (const float*)d_in[9];
    const float* oW    = (const float*)d_in[10];
    const float* obias = (const float*)d_in[11];
    const float* lng   = (const float*)d_in[12];
    const float* lnb   = (const float*)d_in[13];
    float* out = (float*)d_out;

    const int B = 32, N = 1024, M = 32, D = 256, BN = B * N;

    char* w = (char*)d_ws;
    size_t off = 0;
    auto alloc = [&](size_t bytes) -> void* {
        off = (off + 255) & ~(size_t)255;
        void* p = w + off;
        off += bytes;
        return p;
    };

    float*          dinv    = (float*)alloc((size_t)N * 4);
    unsigned short* adjn    = (unsigned short*)alloc((size_t)N * N * 2);
    unsigned short* qWt     = (unsigned short*)alloc((size_t)D * D * 2);
    unsigned short* f1Wt    = (unsigned short*)alloc((size_t)D * 3 * D * 2);
    unsigned short* oWt     = (unsigned short*)alloc((size_t)D * D * 2);
    unsigned short* keysT   = (unsigned short*)alloc((size_t)M * D * N * 2);
    unsigned short* valsT   = (unsigned short*)alloc((size_t)M * D * N * 2);
    unsigned short* cat     = (unsigned short*)alloc((size_t)BN * 3 * D * 2);
    float*          qbuf    = (float*)alloc((size_t)BN * D * 4);   // reused as h
    unsigned short* qn      = (unsigned short*)alloc((size_t)BN * D * 2);
    unsigned short* knS     = (unsigned short*)alloc((size_t)N * M * D * 2);
    float*          nbrK    = (float*)alloc((size_t)N * M * D * 4); // reused as outb
    float*          nbrV    = (float*)alloc((size_t)N * M * D * 4);
    unsigned short* knN     = (unsigned short*)alloc((size_t)N * M * D * 2);
    float*          simS    = (float*)alloc((size_t)BN * M * 4);
    float*          simN    = (float*)alloc((size_t)BN * M * 4);
    float*          selfm   = (float*)alloc((size_t)BN * D * 4);
    float*          nbrm    = (float*)alloc((size_t)BN * D * 4);
    unsigned short* fusedbf = (unsigned short*)alloc((size_t)BN * D * 2);
    float*          h       = qbuf;   // q dead after qn
    float*          outb    = nbrK;   // nbrK dead after knN

    dim3 b256(256);

    // adjacency normalization (bf16 A operand)
    k_dinv   <<<N / 8, b256, 0, stream>>>(adj, dinv, N);
    k_adjnorm<<<(N * N) / 256, b256, 0, stream>>>(adj, dinv, adjn, N);

    // weight / data transposes to bf16 Bt layout
    k_f2bf_T<<<(D * D + 255) / 256, b256, 0, stream>>>(qW, qWt, D, D);
    k_f2bf_T<<<(3 * D * D + 255) / 256, b256, 0, stream>>>(f1W, f1Wt, 3 * D, D);
    k_f2bf_T<<<(D * D + 255) / 256, b256, 0, stream>>>(oW, oWt, D, D);
    k_f2bf_T<<<(int)(((size_t)N * M * D + 255) / 256), b256, 0, stream>>>(mkeys, keysT, N, M * D);
    k_f2bf_T<<<(int)(((size_t)N * M * D + 255) / 256), b256, 0, stream>>>(mvals, valsT, N, M * D);

    // x -> cat[:, 0:256] bf16
    k_f2bf_stride<<<(int)(((size_t)BN * D) / 256), b256, 0, stream>>>(
        x, cat, D, 3 * D, 0, (long long)BN * D);

    // q = x @ qW + qb   (A = cat with lda=768)
    k_gemm<<<dim3(D / 128, BN / 128), b256, 0, stream>>>(cat, 3 * D, qWt, D, qbuf, D, D, qb, 0);
    k_rownorm<<<BN / 8, b256, 0, stream>>>(qbuf, qn, BN);
    k_rownorm<<<(N * M) / 8, b256, 0, stream>>>(mkeys, knS, N * M);

    // nbr_keys / nbr_vals = adj_norm @ mem_{keys,vals}  (the big GEMMs)
    k_gemm<<<dim3((M * D) / 128, N / 128), b256, 0, stream>>>(adjn, N, keysT, N, nbrK, M * D, N, nullptr, 0);
    k_gemm<<<dim3((M * D) / 128, N / 128), b256, 0, stream>>>(adjn, N, valsT, N, nbrV, M * D, N, nullptr, 0);
    k_rownorm<<<(N * M) / 8, b256, 0, stream>>>(nbrK, knN, N * M);

    // similarities (single-wave WMMA per node)
    k_sim<<<N, 32, 0, stream>>>(qn, knS, simS, N);
    k_sim<<<N, 32, 0, stream>>>(qn, knN, simN, N);

    // top-4 read of self / neighbor memories (also fills cat columns)
    k_topk<<<BN / 8, b256, 0, stream>>>(simS, mvals, selfm, cat, D, N);
    k_topk<<<BN / 8, b256, 0, stream>>>(simN, nbrV, nbrm, cat, 2 * D, N);

    // h = gelu(cat @ f1W + f1b)
    k_gemm<<<dim3(D / 128, BN / 128), b256, 0, stream>>>(cat, 3 * D, f1Wt, 3 * D, h, D, 3 * D, f1b, 1);

    // fusion softmax + weighted combine -> fused bf16
    k_fusion<<<BN / 8, b256, 0, stream>>>(h, f2W, f2b, x, selfm, nbrm, fusedbf);

    // out = gelu(fused @ oW + ob)
    k_gemm<<<dim3(D / 128, BN / 128), b256, 0, stream>>>(fusedbf, D, oWt, D, outb, D, D, obias, 1);

    // y = LayerNorm(x + out)
    k_lnres<<<BN / 8, b256, 0, stream>>>(x, outb, lng, lnb, out);
}
